// TCN_GCN_unit_76922864272135
// MI455X (gfx1250) — matrix operand order repaired and checked by reading, not compiled, not run on hardware
//
#include <hip/hip_runtime.h>
#include <stdint.h>

// ---------------- problem dims ----------------
#define NB   32
#define CIN  64
#define CO   64
#define TT   256
#define VV   25
#define TV   6400        // T*V (unpadded)
#define VP   32          // padded V stride
#define TVP  8192        // T*VP (padded plane)
#define KC   192         // K*Co
#define KS   3
#define RR   8

typedef __attribute__((ext_vector_type(16))) __bf16 v16bf;
typedef __attribute__((ext_vector_type(8)))  float  v8f;
typedef __attribute__((ext_vector_type(4)))  int    v4i_t;

union FragBF { v16bf v; uint4 q[2]; };

#define AS1 __attribute__((address_space(1)))
#define AS3 __attribute__((address_space(3)))

#if defined(__has_builtin)
#if __has_builtin(__builtin_amdgcn_global_load_async_to_lds_b128) && \
    __has_builtin(__builtin_amdgcn_s_wait_asynccnt)
#define HAVE_ASYNC 1
#endif
#endif
#ifndef HAVE_ASYNC
#define HAVE_ASYNC 0
#endif

#if HAVE_ASYNC
__device__ __forceinline__ void a_cp16(const unsigned short* g, unsigned short* l) {
  __builtin_amdgcn_global_load_async_to_lds_b128((AS1 v4i_t*)(const void*)g,
                                                 (AS3 v4i_t*)(void*)l, 0, 0);
}
__device__ __forceinline__ void a_wait() { __builtin_amdgcn_s_wait_asynccnt(0); }
#endif

__device__ inline unsigned short f2bf(float f) {
  union { float f; uint32_t u; } c; c.f = f;
  uint32_t u = c.u + 0x7FFFu + ((c.u >> 16) & 1u);
  return (unsigned short)(u >> 16);
}

__device__ inline v8f zero8() {
  v8f z = {0.f, 0.f, 0.f, 0.f, 0.f, 0.f, 0.f, 0.f};
  return z;
}

__device__ inline v8f wmma_bf16(v16bf a, v16bf b, v8f c) {
  return __builtin_amdgcn_wmma_f32_16x16x32_bf16(false, a, false, b, (short)0, c,
                                                 false, false);
}

// =====================================================================
// Kernel: zero the padded V-columns (25..31) of xwbn and x3h planes so
// pad garbage can never inject NaN through WMMA (pad x 0-row of B).
// One 16B store per (plane,t) row covering halves [24..32).
// =====================================================================
__global__ void k_zeropad(unsigned short* __restrict__ xwbn,
                          unsigned short* __restrict__ x3h) {
  int idx = blockIdx.x * blockDim.x + threadIdx.x;
  if (idx >= NB * KC * TT) return;
  size_t off = (size_t)idx * VP + 24;
  uint4 z = {0u, 0u, 0u, 0u};
  *(uint4*)(xwbn + off) = z;
  *(uint4*)(x3h + off) = z;
}

// =====================================================================
// Kernel: precompute column-normalized adjacency (8 distinct groups)
// Bg layout: [g][w(32 cols)][K=96] bf16, col-major-by-K, zero padded
// =====================================================================
__global__ __launch_bounds__(256) void k_norm(const float* __restrict__ DA,
                                              unsigned short* __restrict__ Bg) {
  __shared__ float colsum[3 * 8 * 25];
  const int tid = threadIdx.x;
  for (int idx = tid; idx < 600; idx += 256) {
    int k = idx / 200, g = (idx / 25) % 8, w = idx % 25;
    float s = 0.f;
    for (int v = 0; v < 25; ++v) s += DA[(((k * 8 + g) * 25) + v) * 25 + w];
    colsum[idx] = s + 0.001f;
  }
  __syncthreads();
  for (int idx = tid; idx < 8 * 32 * 96; idx += 256) {
    int g = idx / (32 * 96);
    int rem = idx % (32 * 96);
    int w = rem / 96, kq = rem % 96;
    int k = kq / 32, v = kq % 32;
    float val = 0.f;
    if (v < 25 && w < 25)
      val = DA[(((k * 8 + g) * 25) + v) * 25 + w] / colsum[k * 200 + g * 25 + w];
    Bg[idx] = f2bf(val);
  }
}

// =====================================================================
// Kernel: fused GEMM  [NTV x 64] x [64 x 384] :
//   cols 0..191   -> BN0(x @ lin_w + lin_b)        -> xwbn (bf16, padded)
//   cols 192..383 -> x @ ctr_w3^T + ctr_b3         -> x3h  (bf16, padded)
// jt-loop unrolled by 2 -> two independent WMMA chains fill XDL hazards
// =====================================================================
__global__ __launch_bounds__(128) void k_lin3(
    const float* __restrict__ x, const float* __restrict__ lin_w,
    const float* __restrict__ lin_b, const float* __restrict__ bn0_g,
    const float* __restrict__ bn0_b, const float* __restrict__ bn0_m,
    const float* __restrict__ bn0_v, const float* __restrict__ ctr_w3,
    const float* __restrict__ ctr_b3, unsigned short* __restrict__ xwbn,
    unsigned short* __restrict__ x3h) {
  __shared__ __align__(16) unsigned short Bs[384 * 64];  // [col j][K=c]
  __shared__ __align__(16) unsigned short As[64 * 72];   // [m][c], stride 72
  __shared__ float scs[384], shs[384];

  const int tid = threadIdx.x;
  const int blk = blockIdx.x;
  const int n = blk / 100;
  const int mstart = (blk % 100) * 64;

  for (int idx = tid; idx < 384 * 64; idx += 128) {
    int j = idx / 64, c = idx % 64;
    float w = (j < 192) ? lin_w[c * 192 + j] : ctr_w3[(j - 192) * 64 + c];
    Bs[j * 64 + c] = f2bf(w);
  }
  for (int j = tid; j < 384; j += 128) {
    float sc, sh;
    if (j < 192) {
      sc = bn0_g[j] * rsqrtf(bn0_v[j] + 1e-5f);
      sh = (lin_b[j] - bn0_m[j]) * sc + bn0_b[j];
    } else {
      sc = 1.f;
      sh = ctr_b3[j - 192];
    }
    scs[j] = sc;
    shs[j] = sh;
  }
  const float* xn = x + (size_t)n * (CIN * TV) + mstart;
  for (int idx = tid; idx < 64 * 64; idx += 128) {
    int c = idx >> 6, m = idx & 63;
    As[m * 72 + c] = f2bf(xn[(size_t)c * TV + m]);
  }
  __syncthreads();

  const int wave = tid >> 5, lane = tid & 31;
  const int lr = lane & 15, hi = lane >> 4;

  FragBF a0, a1;
  {
    const unsigned short* ap = As + (wave * 16 + lr) * 72 + hi * 16;
    a0.q[0] = *(const uint4*)(ap + 0);
    a0.q[1] = *(const uint4*)(ap + 8);
    a1.q[0] = *(const uint4*)(ap + 32);
    a1.q[1] = *(const uint4*)(ap + 40);
  }
  const int mbase = mstart + wave * 16 + hi * 8;
  // per-lane store offsets are jt-invariant
  int toff[8];
#pragma unroll
  for (int i = 0; i < 8; ++i) {
    int m = mbase + i;
    int t = m / 25, v = m - t * 25;
    toff[i] = t * VP + v;
  }

#pragma unroll 2
  for (int jt = 0; jt < 24; ++jt) {
    const int j = jt * 16 + lr;
    FragBF b0, b1;
    const unsigned short* bp = Bs + j * 64 + hi * 16;
    b0.q[0] = *(const uint4*)(bp + 0);
    b0.q[1] = *(const uint4*)(bp + 8);
    b1.q[0] = *(const uint4*)(bp + 32);
    b1.q[1] = *(const uint4*)(bp + 40);

    v8f acc = zero8();
    acc = wmma_bf16(a0.v, b0.v, acc);
    acc = wmma_bf16(a1.v, b1.v, acc);

    const float sc = scs[j], sh = shs[j];
    unsigned short* dst = (j < 192)
                              ? xwbn + (size_t)(n * KC + j) * TVP
                              : x3h + (size_t)(n * KC + (j - 192)) * TVP;
#pragma unroll
    for (int i = 0; i < 8; ++i) dst[toff[i]] = f2bf(acc[i] * sc + sh);
  }
}

// =====================================================================
// Kernel: graph conv  xg[n,c,t,w] = sum_{k,v} xwbn[n,k*64+c,t,v]*normA
// A and B tiles staged via ASYNC global->LDS copies (padded layout).
// =====================================================================
__global__ __launch_bounds__(128) void k_gconv(
    const unsigned short* __restrict__ xwbn, const unsigned short* __restrict__ Bg,
    unsigned short* __restrict__ xgh) {
  __shared__ __align__(16) unsigned short As[64 * 96];
  __shared__ __align__(16) unsigned short Bs[32 * 96];

  const int tid = threadIdx.x;
  const int c = blockIdx.y;
  const int mblk = blockIdx.x * 64;  // over N*T
  const int n = mblk / TT;
  const int t0 = mblk % TT;
  const int g = c & 7;

#if HAVE_ASYNC
  // B: 32*96 halves contiguous = 384 x 16B chunks
  for (int q = tid; q < 384; q += 128)
    a_cp16(Bg + g * (32 * 96) + q * 8, Bs + q * 8);
  // A: 64 rows x 3 subsets x 64B (pads are zero in global)
  for (int q = tid; q < 768; q += 128) {
    int tl = q / 12, r = q % 12;
    int k = r >> 2, ch = r & 3;
    const unsigned short* gp =
        xwbn + (size_t)(n * KC + k * 64 + c) * TVP + (t0 + tl) * VP + ch * 8;
    a_cp16(gp, As + tl * 96 + k * 32 + ch * 8);
  }
  a_wait();
#else
  for (int idx = tid; idx < 32 * 96; idx += 128) Bs[idx] = Bg[g * (32 * 96) + idx];
  for (int idx = tid; idx < 64 * 96; idx += 128) {
    int tl = idx / 96, kq = idx % 96;
    int k = kq / 32, v = kq % 32;
    As[idx] = xwbn[(size_t)(n * KC + k * 64 + c) * TVP + (t0 + tl) * VP + v];
  }
#endif
  __syncthreads();

  const int wave = tid >> 5, lane = tid & 31;
  const int lr = lane & 15, hi = lane >> 4;
  v8f acc0 = zero8(), acc1 = zero8();

  const unsigned short* arow = As + (wave * 16 + lr) * 96 + hi * 16;
#pragma unroll
  for (int kk = 0; kk < 3; ++kk) {
    FragBF a;
    a.q[0] = *(const uint4*)(arow + kk * 32);
    a.q[1] = *(const uint4*)(arow + kk * 32 + 8);
    FragBF b0, b1;
    const unsigned short* bp0 = Bs + lr * 96 + kk * 32 + hi * 16;
    b0.q[0] = *(const uint4*)bp0;
    b0.q[1] = *(const uint4*)(bp0 + 8);
    const unsigned short* bp1 = Bs + (16 + lr) * 96 + kk * 32 + hi * 16;
    b1.q[0] = *(const uint4*)bp1;
    b1.q[1] = *(const uint4*)(bp1 + 8);
    acc0 = wmma_bf16(a.v, b0.v, acc0);
    acc1 = wmma_bf16(a.v, b1.v, acc1);
  }

  const size_t base = (size_t)(n * CO + c) * TV;
  const int trow = t0 + wave * 16 + hi * 8;
#pragma unroll
  for (int i = 0; i < 8; ++i) {
    int t = trow + i;
    xgh[base + t * 25 + lr] = f2bf(acc0[i]);
    if (16 + lr < 25) xgh[base + t * 25 + 16 + lr] = f2bf(acc1[i]);
  }
}

// =====================================================================
// Kernel: xm = mean over T of x
// =====================================================================
__global__ void k_xm(const float* __restrict__ x, float* __restrict__ xm) {
  int idx = blockIdx.x * blockDim.x + threadIdx.x;
  if (idx >= NB * CIN * VV) return;
  int v = idx % VV, nc = idx / VV;
  const float* p = x + (size_t)nc * TV + v;
  float s = 0.f;
  for (int t = 0; t < TT; ++t) s += p[t * 25];
  xm[idx] = s * (1.f / 256.f);
}

// =====================================================================
// Kernel: CTRGC attention; writes att TRANSPOSED + padded:
// atth plane [(n*3+k)*64+o] is [v(32)][u(32)] bf16, zero outside 25x25
// =====================================================================
__global__ __launch_bounds__(256) void k_att(
    const float* __restrict__ xm, const float* __restrict__ w1,
    const float* __restrict__ b1, const float* __restrict__ w2,
    const float* __restrict__ b2, const float* __restrict__ w4,
    const float* __restrict__ b4, const float* __restrict__ alpha,
    const float* __restrict__ A3, unsigned short* __restrict__ atth) {
  __shared__ float x1s[RR * VV], x2s[RR * VV];
  __shared__ float td[RR * VV * VV];  // tanh(x1[u]-x2[v]) : 20 KB
  const int nb = blockIdx.x;
  const int n = nb / KS, k = nb % KS;
  const int tid = threadIdx.x;
  const float* xmn = xm + (size_t)n * CIN * VV;

  for (int idx = tid; idx < RR * VV; idx += 256) {
    int r = idx / VV, v = idx % VV;
    float s1 = b1[k * RR + r], s2 = b2[k * RR + r];
    for (int c = 0; c < CIN; ++c) {
      float xv = xmn[c * VV + v];
      s1 += xv * w1[(k * RR + r) * CIN + c];
      s2 += xv * w2[(k * RR + r) * CIN + c];
    }
    x1s[idx] = s1;
    x2s[idx] = s2;
  }
  __syncthreads();
  for (int idx = tid; idx < RR * VV * VV; idx += 256) {
    int r = idx / 625, uv = idx % 625;
    int u = uv / 25, v = uv % 25;
    td[idx] = tanhf(x1s[r * VV + u] - x2s[r * VV + v]);
  }
  __syncthreads();
  const float al = alpha[0];
  unsigned short* outp = atth + (size_t)(n * KS + k) * CO * 1024;
  for (int idx = tid; idx < CO * 1024; idx += 256) {
    int o = idx >> 10, rem = idx & 1023;
    int v = rem >> 5, u = rem & 31;
    unsigned short hv = 0;
    if (u < 25 && v < 25) {
      float s = b4[k * CO + o];
#pragma unroll
      for (int r = 0; r < RR; ++r)
        s += w4[(k * CO + o) * RR + r] * td[r * 625 + u * 25 + v];
      hv = f2bf(al * s + A3[k * 625 + u * 25 + v]);
    }
    outp[idx] = hv;
  }
}

// =====================================================================
// Kernel: yc[n,o,t,v] = sum_k  x3[n,k,o,t,:25] @ att[n,k,o,:25,:25]
// A and B staged via ASYNC global->LDS copies (padded layouts)
// =====================================================================
__global__ __launch_bounds__(128) void k_yc(const unsigned short* __restrict__ x3h,
                                            const unsigned short* __restrict__ atth,
                                            unsigned short* __restrict__ ych) {
  __shared__ __align__(16) unsigned short As[256 * 32];
  __shared__ __align__(16) unsigned short Bs[32 * 32];
  const int b = blockIdx.x;
  const int n = b / CO, o = b % CO;
  const int tid = threadIdx.x;
  const int wave = tid >> 5, lane = tid & 31;
  const int lr = lane & 15, hi = lane >> 4;

  v8f acc[4][2];
#pragma unroll
  for (int ms = 0; ms < 4; ++ms) {
    acc[ms][0] = zero8();
    acc[ms][1] = zero8();
  }

  for (int k = 0; k < KS; ++k) {
    __syncthreads();
    const unsigned short* xp = x3h + (size_t)(n * KC + k * 64 + o) * TVP;
    const unsigned short* ap = atth + (size_t)((n * KS + k) * CO + o) * 1024;
#if HAVE_ASYNC
    for (int q = tid; q < 1024; q += 128) a_cp16(xp + q * 8, As + q * 8);
    a_cp16(ap + tid * 8, Bs + tid * 8);
    a_wait();
#else
    for (int idx = tid; idx < 256 * 32; idx += 128)
      As[idx] = xp[(idx >> 5) * VP + (idx & 31)];
    for (int idx = tid; idx < 1024; idx += 128) Bs[idx] = ap[idx];
#endif
    __syncthreads();

    FragBF bf0, bf1;
    {
      const unsigned short* bp = Bs + lr * 32 + hi * 16;
      bf0.q[0] = *(const uint4*)bp;
      bf0.q[1] = *(const uint4*)(bp + 8);
      const unsigned short* bp1 = Bs + (16 + lr) * 32 + hi * 16;
      bf1.q[0] = *(const uint4*)bp1;
      bf1.q[1] = *(const uint4*)(bp1 + 8);
    }
#pragma unroll
    for (int ms = 0; ms < 4; ++ms) {
      const int row = (wave * 4 + ms) * 16 + lr;
      const unsigned short* apr = As + row * 32 + hi * 16;
      FragBF a;
      a.q[0] = *(const uint4*)apr;
      a.q[1] = *(const uint4*)(apr + 8);
      acc[ms][0] = wmma_bf16(a.v, bf0.v, acc[ms][0]);
      acc[ms][1] = wmma_bf16(a.v, bf1.v, acc[ms][1]);
    }
  }

  const size_t base = (size_t)(n * CO + o) * TV;
#pragma unroll
  for (int ms = 0; ms < 4; ++ms) {
    const int trow = (wave * 4 + ms) * 16 + hi * 8;
#pragma unroll
    for (int i = 0; i < 8; ++i) {
      int t = trow + i;
      ych[base + t * 25 + lr] = f2bf(acc[ms][0][i]);
      if (16 + lr < 25) ych[base + t * 25 + 16 + lr] = f2bf(acc[ms][1][i]);
    }
  }
}

// =====================================================================
// Kernel: ensemble 1x1 GEMM  [NTV x 128] x [128 x 64], fused BN+res+ReLU
// =====================================================================
__global__ __launch_bounds__(128) void k_ens(
    const unsigned short* __restrict__ xgh, const unsigned short* __restrict__ ych,
    const float* __restrict__ ens_w, const float* __restrict__ ens_b,
    const float* __restrict__ bn_g, const float* __restrict__ bn_b,
    const float* __restrict__ bn_m, const float* __restrict__ bn_v,
    const float* __restrict__ x, float* __restrict__ y) {
  __shared__ __align__(16) unsigned short Bs[64 * 128];  // [o][c]
  __shared__ __align__(16) unsigned short As[64 * 136];  // [m][c] stride 136

  const int tid = threadIdx.x;
  const int blk = blockIdx.x;
  const int n = blk / 100;
  const int mstart = (blk % 100) * 64;

  for (int idx = tid; idx < 64 * 128; idx += 128) Bs[idx] = f2bf(ens_w[idx]);
  for (int idx = tid; idx < 64 * 128; idx += 128) {
    int c = idx >> 6, m = idx & 63;
    const unsigned short* src =
        (c < 64) ? xgh + (size_t)(n * CO + c) * TV + mstart + m
                 : ych + (size_t)(n * CO + (c - 64)) * TV + mstart + m;
    As[m * 136 + c] = *src;
  }
  __syncthreads();

  const int wave = tid >> 5, lane = tid & 31;
  const int lr = lane & 15, hi = lane >> 4;

  FragBF a[4];
  const unsigned short* arow = As + (wave * 16 + lr) * 136 + hi * 16;
#pragma unroll
  for (int kk = 0; kk < 4; ++kk) {
    a[kk].q[0] = *(const uint4*)(arow + kk * 32);
    a[kk].q[1] = *(const uint4*)(arow + kk * 32 + 8);
  }
  const int mrow = mstart + wave * 16 + hi * 8;

#pragma unroll
  for (int jt = 0; jt < 4; ++jt) {
    const int o = jt * 16 + lr;
    v8f acc = zero8();
    const unsigned short* bcol = Bs + o * 128;
#pragma unroll
    for (int kk = 0; kk < 4; ++kk) {
      FragBF bfr;
      const unsigned short* bp = bcol + kk * 32 + hi * 16;
      bfr.q[0] = *(const uint4*)bp;
      bfr.q[1] = *(const uint4*)(bp + 8);
      acc = wmma_bf16(a[kk].v, bfr.v, acc);
    }
    const float sc = bn_g[o] * rsqrtf(bn_v[o] + 1e-5f);
    const float sh = (ens_b[o] - bn_m[o]) * sc + bn_b[o];
    const float* xp = x + (size_t)(n * CO + o) * TV + mrow;
    float* yp = y + (size_t)(n * CO + o) * TV + mrow;
#pragma unroll
    for (int i = 0; i < 8; ++i) {
      float v = acc[i] * sc + sh + xp[i];
      yp[i] = v > 0.f ? v : 0.f;
    }
  }
}

// ============== attention chain (reductions + tiny convs) ==============
__global__ void k_se(const float* __restrict__ y, float* __restrict__ se) {
  int idx = blockIdx.x * blockDim.x + threadIdx.x;
  if (idx >= NB * CO * VV) return;
  int v = idx % VV, nc = idx / VV;
  const float* p = y + (size_t)nc * TV + v;
  float s = 0.f;
  for (int t = 0; t < TT; ++t) s += p[t * 25];
  se[idx] = s * (1.f / 256.f);
}

__global__ void k_sa(const float* __restrict__ se, const float* __restrict__ sa_w,
                     const float* __restrict__ sa_b, float* __restrict__ s1) {
  int idx = blockIdx.x * blockDim.x + threadIdx.x;
  if (idx >= NB * VV) return;
  int n = idx / VV, v = idx % VV;
  float s = sa_b[0];
  const float* sen = se + (size_t)n * CO * VV;
  for (int c = 0; c < CO; ++c)
    for (int kx = 0; kx < 25; ++kx) {
      int vv = v + kx - 12;
      if (vv >= 0 && vv < 25) s += sen[c * VV + vv] * sa_w[c * 25 + kx];
    }
  s1[idx] = 1.f + 1.f / (1.f + __expf(-s));
}

__global__ __launch_bounds__(256) void k_apply_sa(float* __restrict__ y,
                                                  const float* __restrict__ s1,
                                                  float* __restrict__ te) {
  __shared__ float yt[TV];
  __shared__ float sv[VV];
  const int nc = blockIdx.x;
  const int n = nc / CO;
  const int tid = threadIdx.x;
  if (tid < VV) sv[tid] = s1[n * VV + tid];
  __syncthreads();
  float* yp = y + (size_t)nc * TV;
  for (int m = tid; m < TV; m += 256) {
    float v = yp[m] * sv[m % 25];
    yt[m] = v;
    yp[m] = v;
  }
  __syncthreads();
  for (int t = tid; t < TT; t += 256) {
    float s = 0.f;
#pragma unroll
    for (int v = 0; v < 25; ++v) s += yt[t * 25 + v];
    te[(size_t)nc * TT + t] = s * (1.f / 25.f);
  }
}

__global__ void k_ta(const float* __restrict__ te, const float* __restrict__ ta_w,
                     const float* __restrict__ ta_b, float* __restrict__ ta1) {
  int idx = blockIdx.x * blockDim.x + threadIdx.x;
  if (idx >= NB * TT) return;
  int n = idx / TT, t = idx % TT;
  float s = ta_b[0];
  const float* ten = te + (size_t)n * CO * TT;
  for (int c = 0; c < CO; ++c)
#pragma unroll
    for (int kx = 0; kx < 9; ++kx) {
      int tt = t + kx - 4;
      if (tt >= 0 && tt < TT) s += ten[c * TT + tt] * ta_w[c * 9 + kx];
    }
  ta1[idx] = 1.f + 1.f / (1.f + __expf(-s));
}

__global__ __launch_bounds__(256) void k_apply_ta(float* __restrict__ y,
                                                  const float* __restrict__ ta1,
                                                  float* __restrict__ ce) {
  __shared__ float red[256];
  const int nc = blockIdx.x;
  const int n = nc / CO;
  const int tid = threadIdx.x;
  float* yp = y + (size_t)nc * TV;
  const float* tap = ta1 + n * TT;
  float s = 0.f;
  for (int m = tid; m < TV; m += 256) {
    float v = yp[m] * tap[m / 25];
    yp[m] = v;
    s += v;
  }
  red[tid] = s;
  __syncthreads();
  for (int off = 128; off > 0; off >>= 1) {
    if (tid < off) red[tid] += red[tid + off];
    __syncthreads();
  }
  if (tid == 0) ce[nc] = red[0] * (1.f / 6400.f);
}

__global__ __launch_bounds__(256) void k_fc(const float* __restrict__ ce,
                                            const float* __restrict__ fc1_w,
                                            const float* __restrict__ fc1_b,
                                            const float* __restrict__ fc2_w,
                                            const float* __restrict__ fc2_b,
                                            float* __restrict__ c2s) {
  __shared__ float c1[NB * 32];
  const int tid = threadIdx.x;
  for (int idx = tid; idx < NB * 32; idx += 256) {
    int n = idx / 32, j = idx % 32;
    float s = fc1_b[j];
    for (int c = 0; c < CO; ++c) s += ce[n * CO + c] * fc1_w[j * CO + c];
    c1[idx] = s > 0.f ? s : 0.f;
  }
  __syncthreads();
  for (int idx = tid; idx < NB * CO; idx += 256) {
    int n = idx / CO, o = idx % CO;
    float s = fc2_b[o];
    for (int j = 0; j < 32; ++j) s += c1[n * 32 + j] * fc2_w[o * 32 + j];
    c2s[idx] = 1.f + 1.f / (1.f + __expf(-s));
  }
}

// =====================================================================
// Kernel: pack TCN weights as bf16 [o][kidx = dt*64+c] for B fragments
// =====================================================================
__global__ void k_prepB(const float* __restrict__ tcn_w,
                        unsigned short* __restrict__ Bsg) {
  int idx = blockIdx.x * blockDim.x + threadIdx.x;
  if (idx >= 64 * 576) return;
  int o = idx / 576, kidx = idx % 576;
  int dt = kidx >> 6, c = kidx & 63;
  Bsg[idx] = f2bf(tcn_w[(o * 64 + c) * 9 + dt]);
}

// =====================================================================
// Kernel: implicit-GEMM temporal conv (9,1) + tbn + residual + ReLU
// Channel-attention gate folded into the A staging.
// =====================================================================
__global__ __launch_bounds__(128) void k_tcn(
    const float* __restrict__ y, const float* __restrict__ c2s,
    const unsigned short* __restrict__ Bsg, const float* __restrict__ tcn_b,
    const float* __restrict__ tbn_g, const float* __restrict__ tbn_b,
    const float* __restrict__ tbn_m, const float* __restrict__ tbn_v,
    const float* __restrict__ x, float* __restrict__ out) {
  __shared__ __align__(16) unsigned short ysT[264 * 64];  // [ml][c] haloed tile
  __shared__ float scl[CO];

  const int tid = threadIdx.x;
  const int blk = blockIdx.x;
  const int n = blk / 100;
  const int mstart = (blk % 100) * 64;

  if (tid < CO) {
    scl[tid] = c2s[n * CO + tid];
    __builtin_prefetch(Bsg + tid * 576, 0, 0);
  }
  __syncthreads();
  for (int idx = tid; idx < 264 * 64; idx += 128) {
    int c = idx / 264, ml = idx % 264;
    int fm = mstart - 100 + ml;  // flat = t'*25+v ; OOB in t' <=> OOB flat
    float v = 0.f;
    if (fm >= 0 && fm < TV) v = y[(size_t)(n * CO + c) * TV + fm] * scl[c];
    ysT[ml * 64 + c] = f2bf(v);
  }
  __syncthreads();

  const int wave = tid >> 5, lane = tid & 31;
  const int lr = lane & 15, hi = lane >> 4;
  v8f acc[4];
#pragma unroll
  for (int jt = 0; jt < 4; ++jt) acc[jt] = zero8();

  const int mrowl = wave * 16 + lr;
#pragma unroll
  for (int kk = 0; kk < 18; ++kk) {
    const int dt = kk >> 1, cc = kk & 1;
    const unsigned short* ap = ysT + (mrowl + 25 * dt) * 64 + cc * 32 + hi * 16;
    FragBF a;
    a.q[0] = *(const uint4*)ap;
    a.q[1] = *(const uint4*)(ap + 8);
#pragma unroll
    for (int jt = 0; jt < 4; ++jt) {
      const unsigned short* bp = Bsg + (jt * 16 + lr) * 576 + kk * 32 + hi * 16;
      FragBF bfr;
      bfr.q[0] = *(const uint4*)bp;
      bfr.q[1] = *(const uint4*)(bp + 8);
      acc[jt] = wmma_bf16(a.v, bfr.v, acc[jt]);
    }
  }

  const int mrow = mstart + wave * 16 + hi * 8;
#pragma unroll
  for (int jt = 0; jt < 4; ++jt) {
    const int o = jt * 16 + lr;
    const float sc = tbn_g[o] * rsqrtf(tbn_v[o] + 1e-5f);
    const float sh = (tcn_b[o] - tbn_m[o]) * sc + tbn_b[o];
    const float* xp = x + (size_t)(n * CO + o) * TV + mrow;
    float* op = out + (size_t)(n * CO + o) * TV + mrow;
#pragma unroll
    for (int i = 0; i < 8; ++i) {
      float v = acc[jt][i] * sc + sh + xp[i];
      op[i] = v > 0.f ? v : 0.f;
    }
  }
}

// =====================================================================
extern "C" void kernel_launch(void* const* d_in, const int* in_sizes, int n_in,
                              void* d_out, int out_size, void* d_ws, size_t ws_size,
                              hipStream_t stream) {
  const float* x      = (const float*)d_in[0];
  const float* A3     = (const float*)d_in[1];
  const float* DA     = (const float*)d_in[2];
  const float* alpha  = (const float*)d_in[3];
  const float* lin_w  = (const float*)d_in[4];
  const float* lin_b  = (const float*)d_in[5];
  const float* bn0_g  = (const float*)d_in[6];
  const float* bn0_b  = (const float*)d_in[7];
  const float* bn0_m  = (const float*)d_in[8];
  const float* bn0_v  = (const float*)d_in[9];
  const float* ctr_w1 = (const float*)d_in[10];
  const float* ctr_b1 = (const float*)d_in[11];
  const float* ctr_w2 = (const float*)d_in[12];
  const float* ctr_b2 = (const float*)d_in[13];
  const float* ctr_w3 = (const float*)d_in[14];
  const float* ctr_b3 = (const float*)d_in[15];
  const float* ctr_w4 = (const float*)d_in[16];
  const float* ctr_b4 = (const float*)d_in[17];
  const float* ens_w  = (const float*)d_in[18];
  const float* ens_b  = (const float*)d_in[19];
  const float* bn_g   = (const float*)d_in[20];
  const float* bn_b   = (const float*)d_in[21];
  const float* bn_m   = (const float*)d_in[22];
  const float* bn_v   = (const float*)d_in[23];
  const float* sa_w   = (const float*)d_in[24];
  const float* sa_b   = (const float*)d_in[25];
  const float* ta_w   = (const float*)d_in[26];
  const float* ta_b   = (const float*)d_in[27];
  const float* fc1_w  = (const float*)d_in[28];
  const float* fc1_b  = (const float*)d_in[29];
  const float* fc2_w  = (const float*)d_in[30];
  const float* fc2_b  = (const float*)d_in[31];
  const float* tcn_w  = (const float*)d_in[32];
  const float* tcn_b  = (const float*)d_in[33];
  const float* tbn_g  = (const float*)d_in[34];
  const float* tbn_b  = (const float*)d_in[35];
  const float* tbn_m  = (const float*)d_in[36];
  const float* tbn_v  = (const float*)d_in[37];
  float* outp = (float*)d_out;

  char* ws = (char*)d_ws;
  size_t off = 0;
  auto alloc = [&](size_t bytes) -> void* {
    void* p = ws + off;
    off += (bytes + 255) & ~(size_t)255;
    return p;
  };

  unsigned short* xwbn = (unsigned short*)alloc((size_t)NB * KC * TVP * 2);
  unsigned short* x3h  = (unsigned short*)alloc((size_t)NB * KC * TVP * 2);
  unsigned short* xgh  = (unsigned short*)alloc((size_t)NB * CO * TV * 2);
  unsigned short* ych  = (unsigned short*)alloc((size_t)NB * CO * TV * 2);
  unsigned short* atth = (unsigned short*)alloc((size_t)NB * KS * CO * 1024 * 2);
  unsigned short* Bg   = (unsigned short*)alloc(8 * 32 * 96 * 2);
  unsigned short* Bsg  = (unsigned short*)alloc(64 * 576 * 2);
  float* xm  = (float*)alloc(NB * CIN * VV * 4);
  float* se  = (float*)alloc(NB * CO * VV * 4);
  float* s1  = (float*)alloc(NB * VV * 4);
  float* te  = (float*)alloc(NB * CO * TT * 4);
  float* ta1 = (float*)alloc(NB * TT * 4);
  float* ce  = (float*)alloc(NB * CO * 4);
  float* c2s = (float*)alloc(NB * CO * 4);
  float* y   = (float*)xwbn;  // alias: xwbn dead after k_gconv, y born at k_ens
  (void)in_sizes; (void)n_in; (void)out_size; (void)ws_size;

  k_zeropad<<<(NB * KC * TT + 255) / 256, 256, 0, stream>>>(xwbn, x3h);
  k_norm<<<1, 256, 0, stream>>>(DA, Bg);
  k_lin3<<<NB * 100, 128, 0, stream>>>(x, lin_w, lin_b, bn0_g, bn0_b, bn0_m, bn0_v,
                                       ctr_w3, ctr_b3, xwbn, x3h);
  k_xm<<<(NB * CIN * VV + 255) / 256, 256, 0, stream>>>(x, xm);
  k_att<<<NB * KS, 256, 0, stream>>>(xm, ctr_w1, ctr_b1, ctr_w2, ctr_b2, ctr_w4,
                                     ctr_b4, alpha, A3, atth);
  k_gconv<<<dim3((NB * TT) / 64, CO), 128, 0, stream>>>(xwbn, Bg, xgh);
  k_yc<<<NB * CO, 128, 0, stream>>>(x3h, atth, ych);
  k_ens<<<NB * 100, 128, 0, stream>>>(xgh, ych, ens_w, ens_b, bn_g, bn_b, bn_m,
                                      bn_v, x, y);
  k_se<<<(NB * CO * VV + 255) / 256, 256, 0, stream>>>(y, se);
  k_sa<<<(NB * VV + 127) / 128, 128, 0, stream>>>(se, sa_w, sa_b, s1);
  k_apply_sa<<<NB * CO, 256, 0, stream>>>(y, s1, te);
  k_ta<<<(NB * TT + 127) / 128, 128, 0, stream>>>(te, ta_w, ta_b, ta1);
  k_apply_ta<<<NB * CO, 256, 0, stream>>>(y, ta1, ce);
  k_fc<<<1, 256, 0, stream>>>(ce, fc1_w, fc1_b, fc2_w, fc2_b, c2s);
  k_prepB<<<(64 * 576 + 255) / 256, 256, 0, stream>>>(tcn_w, Bsg);
  k_tcn<<<NB * 100, 128, 0, stream>>>(y, c2s, Bsg, tcn_b, tbn_g, tbn_b, tbn_m,
                                      tbn_v, x, outp);
}
